// AmaFilter_35158602285279
// MI455X (gfx1250) — matrix operand order repaired and checked by reading, not compiled, not run on hardware
//
#include <hip/hip_runtime.h>
#include <math.h>

#define NPTS   2048
#define BCL    8
#define NTOT   (NPTS * BCL)      // 16384 points
#define KNBR   16
#define HIDW   128
#define EDGES  (NTOT * KNBR)     // 262144
#define BN_EPS 1e-5f

typedef __attribute__((ext_vector_type(16))) _Float16 v16h;
typedef __attribute__((ext_vector_type(8)))  _Float16 v8h;
typedef __attribute__((ext_vector_type(8)))  float    v8f;

union V16U { v16h v; v8h h[2]; };

__device__ __forceinline__ float warp_sum32(float s) {
    for (int off = 16; off > 0; off >>= 1) s += __shfl_xor(s, off, 32);
    return s;
}

// ---------------------------------------------------------------------------
// Elementwise / staging kernels
// ---------------------------------------------------------------------------
__global__ void zero_f32(float* p, int n) {
    int i = blockIdx.x * blockDim.x + threadIdx.x;
    if (i < n) p[i] = 0.0f;
}

// dst[r*Kp + k] = (k<K) ? f16(src[r*K + k]) : 0   (pad K to multiple of 32)
__global__ void f32_to_f16_pad(const float* __restrict__ src, _Float16* __restrict__ dst,
                               int rows, int K, int Kp) {
    int i = blockIdx.x * blockDim.x + threadIdx.x;
    int tot = rows * Kp;
    if (i >= tot) return;
    int r = i / Kp, k = i - r * Kp;
    dst[i] = (k < K) ? (_Float16)src[r * K + k] : (_Float16)0.0f;
}

// W[K x Nn] (row-major) -> Wt[NnP x Kp] f16, zero-padded in both K and N.
// N-padding lets the GEMM load B fragments unconditionally (no EXEC dance).
__global__ void w_to_f16_t_pad(const float* __restrict__ W, _Float16* __restrict__ dst,
                               int K, int Nn, int NnP, int Kp) {
    int i = blockIdx.x * blockDim.x + threadIdx.x;
    int tot = NnP * Kp;
    if (i >= tot) return;
    int n = i / Kp, k = i - n * Kp;
    dst[i] = (k < K && n < Nn) ? (_Float16)W[k * Nn + n] : (_Float16)0.0f;
}

__global__ void compute_sq(const float* __restrict__ h, float* __restrict__ sq, int F) {
    int n = blockIdx.x * blockDim.x + threadIdx.x;
    if (n >= NTOT) return;
    float s = 0.0f;
    for (int f = 0; f < F; ++f) { float v = h[n * F + f]; s += v * v; }
    sq[n] = s;
}

// ---------------------------------------------------------------------------
// WMMA GEMM:  C[M x Nn] = A[M x Kp](f16, padded) @ Bt[NnP x Kp]^T + bias
// 64x64 block tile, 128 threads (4 waves). Each wave owns one 16-wide N strip
// and 4 M-subtiles: the B fragment is reused across 4 WMMAs per K-step.
// M must be a multiple of 64.
// ---------------------------------------------------------------------------
__global__ void wmma_gemm_bias(const _Float16* __restrict__ A,
                               const _Float16* __restrict__ Bt,
                               const float* __restrict__ bias,
                               float* __restrict__ C,
                               int M, int Nn, int NnP, int Kp) {
    const int wave = threadIdx.x >> 5;
    const int lane = threadIdx.x & 31;
    const int half = lane >> 4;
    const int ml   = lane & 15;
    const int m0 = blockIdx.x * 64;
    const int n0 = (blockIdx.y * 4 + wave) * 16;
    if (n0 >= NnP) return;                      // wave-uniform exit
    const int n = n0 + ml;

    v8f acc[4] = {{}, {}, {}, {}};
    const _Float16* bptr = Bt + (size_t)n * Kp + half * 16;
    const _Float16* aptr = A + (size_t)(m0 + ml) * Kp + half * 16;
    const size_t subStride = (size_t)16 * Kp;   // 16 rows of A per subtile

    for (int k0 = 0; k0 < Kp; k0 += 32) {
        const v16h b = *(const v16h*)(bptr + k0);
#pragma unroll
        for (int sub = 0; sub < 4; ++sub) {
            V16U a;
            const _Float16* ap = aptr + sub * subStride + k0;
            a.h[0] = *(const v8h*)(ap);
            a.h[1] = *(const v8h*)(ap + 8);
            acc[sub] = __builtin_amdgcn_wmma_f32_16x16x32_f16(false, a.v, false, b,
                                                              (short)0, acc[sub], false, false);
        }
    }
    if (n < Nn) {
        const float bv = bias ? bias[n] : 0.0f;
        const int mrow = half * 8;
#pragma unroll
        for (int sub = 0; sub < 4; ++sub) {
#pragma unroll
            for (int v = 0; v < 8; ++v) {
                int mm = m0 + sub * 16 + mrow + v;
                C[(size_t)mm * Nn + n] = acc[sub][v] + bv;
            }
        }
    }
}

// ---------------------------------------------------------------------------
// Fused kNN: WMMA Gram tiles + streaming top-16 selection per row.
// Templated on padded feature width so the preloaded A fragments stay in
// registers (fully unrolled loops, no dynamic indexing -> no scratch).
// grid.x = BCL * (NPTS/16), block = 128 (4 waves).
// ---------------------------------------------------------------------------
template <int FP>
__global__ void knn_kernel(const _Float16* __restrict__ Xh, const float* __restrict__ sq,
                           int* __restrict__ idx) {
    constexpr int NKS = FP / 32;
    const int tiles_per_cloud = NPTS / 16;
    const int b  = blockIdx.x / tiles_per_cloud;
    const int mt = blockIdx.x % tiles_per_cloud;
    const int wave = threadIdx.x >> 5;
    const int lane = threadIdx.x & 31;
    const int half = lane >> 4;
    const int mlocal = lane & 15;
    const int rbase = b * NPTS + mt * 16;

    __shared__ float sD[16][64];
    __shared__ float ssq[16];
    __shared__ float bestv[16][KNBR];
    __shared__ int   besti[16][KNBR];

    for (int i = threadIdx.x; i < 16 * KNBR; i += blockDim.x) {
        bestv[i / KNBR][i % KNBR] = 3.0e38f;
        besti[i / KNBR][i % KNBR] = 0;
    }
    if (threadIdx.x < 16) ssq[threadIdx.x] = sq[rbase + threadIdx.x];
    __syncthreads();

    // Preload the 16-row A tile fragments once; reused across all 32 chunks.
    V16U afrag[NKS];
    {
        const _Float16* arow = Xh + (size_t)(rbase + mlocal) * FP + half * 16;
#pragma unroll
        for (int ks = 0; ks < NKS; ++ks) {
            afrag[ks].h[0] = *(const v8h*)(arow + ks * 32);
            afrag[ks].h[1] = *(const v8h*)(arow + ks * 32 + 8);
        }
    }

    for (int c0 = 0; c0 < NPTS; c0 += 64) {
        const int n0 = c0 + wave * 16;
        const int colpt = b * NPTS + n0 + mlocal;
        const _Float16* brow = Xh + (size_t)colpt * FP + half * 16;
        v8f acc = {};
#pragma unroll
        for (int ks = 0; ks < NKS; ++ks) {
            v16h bb = *(const v16h*)(brow + ks * 32);
            acc = __builtin_amdgcn_wmma_f32_16x16x32_f16(false, afrag[ks].v, false, bb,
                                                         (short)0, acc, false, false);
        }
        const float sqc = sq[colpt];
#pragma unroll
        for (int v = 0; v < 8; ++v) {
            int mloc = half * 8 + v;
            float d = ssq[mloc] + sqc - 2.0f * acc[v];
            if (rbase + mloc == colpt) d += 1e10f;       // exclude self-loop
            sD[mloc][wave * 16 + mlocal] = d;
        }
        __syncthreads();
        if (threadIdx.x < 16) {                          // streaming top-K insert
            const int m = threadIdx.x;
            for (int j = 0; j < 64; ++j) {
                float dv = sD[m][j];
                if (dv < bestv[m][KNBR - 1]) {
                    int pos = KNBR - 1;
                    while (pos > 0 && bestv[m][pos - 1] > dv) {
                        bestv[m][pos] = bestv[m][pos - 1];
                        besti[m][pos] = besti[m][pos - 1];
                        --pos;
                    }
                    bestv[m][pos] = dv;
                    besti[m][pos] = b * NPTS + c0 + j;   // global neighbor id
                }
            }
        }
        __syncthreads();
    }
    if (threadIdx.x < 16) {
        const int m = threadIdx.x;
        for (int kk = 0; kk < KNBR; ++kk)
            idx[(size_t)(rbase + m) * KNBR + kk] = besti[m][kk];
    }
}

// ---------------------------------------------------------------------------
// Graph kernels
// ---------------------------------------------------------------------------
__global__ void count_rows(const int* __restrict__ idx, float* __restrict__ cnt) {
    int e = blockIdx.x * blockDim.x + threadIdx.x;
    if (e < EDGES) atomicAdd(&cnt[idx[e]], 1.0f);
}

// per-column weighted stats: mean[j], biased var[j] over rows (weight = wgt or 1)
__global__ void colstats(const float* __restrict__ H, int C, int rows,
                         const float* __restrict__ wgt, float wsum,
                         float* __restrict__ mean, float* __restrict__ var) {
    const int j = blockIdx.x;
    float s1 = 0.0f, s2 = 0.0f;
    for (int n = threadIdx.x; n < rows; n += blockDim.x) {
        float h = H[(size_t)n * C + j];
        float w = wgt ? wgt[n] : 1.0f;
        s1 += w * h;
        s2 += w * h * h;
    }
    __shared__ float r1[8], r2[8];
    s1 = warp_sum32(s1);
    s2 = warp_sum32(s2);
    if ((threadIdx.x & 31) == 0) { r1[threadIdx.x >> 5] = s1; r2[threadIdx.x >> 5] = s2; }
    __syncthreads();
    if (threadIdx.x == 0) {
        float t1 = 0.0f, t2 = 0.0f;
        for (int w = 0; w < (int)(blockDim.x >> 5); ++w) { t1 += r1[w]; t2 += r2[w]; }
        float m = t1 / wsum;
        mean[j] = m;
        var[j]  = t2 / wsum - m * m;
    }
}

__global__ void bn_apply(const float* __restrict__ H, float* __restrict__ O,
                         int C, int rows,
                         const float* __restrict__ mean, const float* __restrict__ var,
                         const float* __restrict__ g, const float* __restrict__ be,
                         int relu) {
    int i = blockIdx.x * blockDim.x + threadIdx.x;
    int tot = rows * C;
    if (i >= tot) return;
    int c = i % C;
    float y = g[c] * (H[i] - mean[c]) * rsqrtf(var[c] + BN_EPS) + be[c];
    if (relu) y = fmaxf(y, 0.0f);
    O[i] = y;
}

// one wave per center node: edge weights + degree (col-grouped, no atomics)
__global__ void edge_w_deg(const float* __restrict__ f1n, const float* __restrict__ f2n,
                           const int* __restrict__ idx, float* __restrict__ wbuf,
                           float* __restrict__ deg, int F) {
    const int wave = threadIdx.x >> 5;
    const int lane = threadIdx.x & 31;
    const int c = blockIdx.x * (blockDim.x >> 5) + wave;
    if (c >= NTOT) return;
    float degacc = 0.0f;
    for (int kk = 0; kk < KNBR; ++kk) {
        const int r = idx[(size_t)c * KNBR + kk];
        float s = 0.0f;
        for (int f = lane; f < F; f += 32) {
            float d = f1n[(size_t)r * F + f] - f2n[(size_t)c * F + f];
            s += d * d;
        }
        s = warp_sum32(s);                    // full dist^2 on every lane
        float w = expf(-s) + 1e-9f;
        if (lane == 0) wbuf[(size_t)c * KNBR + kk] = w;
        degacc += w;
    }
    if (lane == 0) deg[c] = degacc;
}

// one wave per center: out[c] = sum_k clip(w/deg[r] * xp[r], +/-10)
__global__ void aggregate(const float* __restrict__ xp, const float* __restrict__ wbuf,
                          const float* __restrict__ deg, const int* __restrict__ idx,
                          float* __restrict__ out, int F) {
    const int wave = threadIdx.x >> 5;
    const int lane = threadIdx.x & 31;
    const int c = blockIdx.x * (blockDim.x >> 5) + wave;
    if (c >= NTOT) return;
    int   rarr[KNBR];
    float sarr[KNBR];
    for (int kk = 0; kk < KNBR; ++kk) {
        rarr[kk] = idx[(size_t)c * KNBR + kk];
        sarr[kk] = wbuf[(size_t)c * KNBR + kk] / deg[rarr[kk]];
    }
    for (int f = lane; f < F; f += 32) {
        float acc = 0.0f;
        for (int kk = 0; kk < KNBR; ++kk) {
            float v = sarr[kk] * xp[(size_t)rarr[kk] * F + f];
            v = fminf(fmaxf(v, -10.0f), 10.0f);
            acc += v;
        }
        out[(size_t)c * F + f] = acc;
    }
}

__global__ void concat_prelu(const float* __restrict__ hold, int fin,
                             const float* __restrict__ out, int fo,
                             const float* __restrict__ aslope,
                             float* __restrict__ dst) {
    const int C = fin + fo;
    int i = blockIdx.x * blockDim.x + threadIdx.x;
    int tot = NTOT * C;
    if (i >= tot) return;
    int n = i / C, c = i - n * C;
    float v = (c < fin) ? hold[(size_t)n * fin + c] : out[(size_t)n * fo + (c - fin)];
    float a = aslope[0];
    dst[i] = (v >= 0.0f) ? v : a * v;
}

__global__ void mse_partial(const float* __restrict__ h, const float* __restrict__ y,
                            int n, float* __restrict__ acc) {
    float s = 0.0f;
    for (int i = blockIdx.x * blockDim.x + threadIdx.x; i < n; i += gridDim.x * blockDim.x) {
        float d = h[i] - y[i];
        s += d * d;
    }
    s = warp_sum32(s);
    __shared__ float sh[8];
    if ((threadIdx.x & 31) == 0) sh[threadIdx.x >> 5] = s;
    __syncthreads();
    if (threadIdx.x == 0) {
        float t = 0.0f;
        for (int w = 0; w < (int)(blockDim.x >> 5); ++w) t += sh[w];
        atomicAdd(acc, t);
    }
}

__global__ void mse_final(const float* __restrict__ acc, float* __restrict__ out2, float inv) {
    if (threadIdx.x == 0) { out2[0] = acc[0] * inv; out2[1] = acc[0] * inv; }
}

// ---------------------------------------------------------------------------
// Host orchestration
// ---------------------------------------------------------------------------
static inline int ceil_div(int a, int b) { return (a + b - 1) / b; }
static inline int pad16(int x) { return (x + 15) & ~15; }
static inline int pad32(int x) { return (x + 31) & ~31; }

extern "C" void kernel_launch(void* const* d_in, const int* in_sizes, int n_in,
                              void* d_out, int out_size, void* d_ws, size_t ws_size,
                              hipStream_t stream) {
    (void)in_sizes; (void)n_in; (void)out_size; (void)ws_size;

    const float* x = (const float*)d_in[0];   // [N,6]
    const float* y = (const float*)d_in[1];   // [N,6]
    // JAX pytree flattening sorts dict keys:
    // params: act0(a,be,g) act1(a,be,g) filt0..2(W1,W2,Wp,b1,b2,be1,be2,bp,g1,g2)
    const float* act_a[2]  = { (const float*)d_in[2], (const float*)d_in[5] };
    const float* act_be[2] = { (const float*)d_in[3], (const float*)d_in[6] };
    const float* act_g[2]  = { (const float*)d_in[4], (const float*)d_in[7] };
    const int fbase[3] = { 8, 18, 28 };
    const float *W1[3], *W2[3], *Wp[3], *b1[3], *b2[3], *be1[3], *be2[3], *bp[3], *g1[3], *g2[3];
    for (int l = 0; l < 3; ++l) {
        W1[l]  = (const float*)d_in[fbase[l] + 0];
        W2[l]  = (const float*)d_in[fbase[l] + 1];
        Wp[l]  = (const float*)d_in[fbase[l] + 2];
        b1[l]  = (const float*)d_in[fbase[l] + 3];
        b2[l]  = (const float*)d_in[fbase[l] + 4];
        be1[l] = (const float*)d_in[fbase[l] + 5];
        be2[l] = (const float*)d_in[fbase[l] + 6];
        bp[l]  = (const float*)d_in[fbase[l] + 7];
        g1[l]  = (const float*)d_in[fbase[l] + 8];
        g2[l]  = (const float*)d_in[fbase[l] + 9];
    }
    const int finA[3] = { 6, 70, 198 };
    const int foA[3]  = { 64, 128, 6 };

    // Workspace carve (bump allocator, 256B aligned)
    char* wp = (char*)d_ws;
    auto alloc = [&](size_t bytes) -> void* {
        void* r = (void*)wp;
        wp += (bytes + 255) & ~(size_t)255;
        return r;
    };
    float*    hA    = (float*)alloc((size_t)NTOT * 256 * 4);
    float*    hB    = (float*)alloc((size_t)NTOT * 256 * 4);
    float*    sq    = (float*)alloc((size_t)NTOT * 4);
    int*      idx   = (int*)  alloc((size_t)EDGES * 4);
    _Float16* xh    = (_Float16*)alloc((size_t)NTOT * 224 * 2);
    _Float16* wt    = (_Float16*)alloc((size_t)128 * 224 * 2);
    float*    xp    = (float*)alloc((size_t)NTOT * 128 * 4);
    float*    h1    = (float*)alloc((size_t)NTOT * 128 * 4);
    float*    erow  = (float*)alloc((size_t)NTOT * 128 * 4);
    float*    ecol  = (float*)alloc((size_t)NTOT * 128 * 4);
    float*    f1n   = (float*)alloc((size_t)NTOT * 128 * 4);
    float*    f2n   = (float*)alloc((size_t)NTOT * 128 * 4);
    float*    wbuf  = (float*)alloc((size_t)EDGES * 4);
    float*    deg   = (float*)alloc((size_t)NTOT * 4);
    float*    cnt   = (float*)alloc((size_t)NTOT * 4);
    float*    mR    = (float*)alloc(256 * 4);
    float*    vR    = (float*)alloc(256 * 4);
    float*    mC    = (float*)alloc(256 * 4);
    float*    vC    = (float*)alloc(256 * 4);
    float*    acc   = (float*)alloc(4 * 4);
    float*    hout  = (float*)d_out;              // final h [N,6] lives in d_out

    const int T = 256;
    auto g1d = [&](int n) { return dim3(ceil_div(n, T)); };
    auto gemm = [&](const _Float16* A, const _Float16* Bt, const float* bias, float* C,
                    int Nn, int NnP, int Kp) {
        wmma_gemm_bias<<<dim3(NTOT / 64, ceil_div(NnP, 64)), 128, 0, stream>>>(
            A, Bt, bias, C, NTOT, Nn, NnP, Kp);
    };

    const float* hcur = x;      // layer input (width fin)
    float* hnext = hA;

    for (int l = 0; l < 3; ++l) {
        const int F  = finA[l], FO = foA[l];
        const int Fp = pad32(F), FOp = pad32(FO), FOn = pad16(FO);

        // ---- stage features to f16 (padded), squared norms, kNN ----
        f32_to_f16_pad<<<g1d(NTOT * Fp), T, 0, stream>>>(hcur, xh, NTOT, F, Fp);
        compute_sq<<<g1d(NTOT), T, 0, stream>>>(hcur, sq, F);
        if (Fp == 32)       knn_kernel<32 ><<<dim3(BCL * (NPTS / 16)), 128, 0, stream>>>(xh, sq, idx);
        else if (Fp == 96)  knn_kernel<96 ><<<dim3(BCL * (NPTS / 16)), 128, 0, stream>>>(xh, sq, idx);
        else                knn_kernel<224><<<dim3(BCL * (NPTS / 16)), 128, 0, stream>>>(xh, sq, idx);

        // ---- fproj: xp = h @ Wp + bp  (WMMA) ----
        w_to_f16_t_pad<<<g1d(FOn * Fp), T, 0, stream>>>(Wp[l], wt, F, FO, FOn, Fp);
        gemm(xh, wt, bp[l], xp, FO, FOn, Fp);

        // ---- in-degree counts (row multiset weights for BN stats) ----
        zero_f32<<<g1d(NTOT), T, 0, stream>>>(cnt, NTOT);
        count_rows<<<g1d(EDGES), T, 0, stream>>>(idx, cnt);

        // ---- embedding layer 1: h1 = xp @ W1 + b1, BN(row/col) + ReLU ----
        f32_to_f16_pad<<<g1d(NTOT * FOp), T, 0, stream>>>(xp, xh, NTOT, FO, FOp);
        w_to_f16_t_pad<<<g1d(HIDW * FOp), T, 0, stream>>>(W1[l], wt, FO, HIDW, HIDW, FOp);
        gemm(xh, wt, b1[l], h1, HIDW, HIDW, FOp);
        colstats<<<dim3(HIDW), T, 0, stream>>>(h1, HIDW, NTOT, cnt, (float)EDGES, mR, vR);
        colstats<<<dim3(HIDW), T, 0, stream>>>(h1, HIDW, NTOT, nullptr, (float)NTOT, mC, vC);
        bn_apply<<<g1d(NTOT * HIDW), T, 0, stream>>>(h1, erow, HIDW, NTOT, mR, vR, g1[l], be1[l], 1);
        bn_apply<<<g1d(NTOT * HIDW), T, 0, stream>>>(h1, ecol, HIDW, NTOT, mC, vC, g1[l], be1[l], 1);

        // ---- embedding layer 2 (row path): f1n ----
        w_to_f16_t_pad<<<g1d(FOn * HIDW), T, 0, stream>>>(W2[l], wt, HIDW, FO, FOn, HIDW);
        f32_to_f16_pad<<<g1d(NTOT * HIDW), T, 0, stream>>>(erow, xh, NTOT, HIDW, HIDW);
        gemm(xh, wt, b2[l], h1, FO, FOn, HIDW);
        colstats<<<dim3(FO), T, 0, stream>>>(h1, FO, NTOT, cnt, (float)EDGES, mR, vR);
        bn_apply<<<g1d(NTOT * FO), T, 0, stream>>>(h1, f1n, FO, NTOT, mR, vR, g2[l], be2[l], 1);

        // ---- embedding layer 2 (col path): f2n ----
        f32_to_f16_pad<<<g1d(NTOT * HIDW), T, 0, stream>>>(ecol, xh, NTOT, HIDW, HIDW);
        gemm(xh, wt, b2[l], h1, FO, FOn, HIDW);
        colstats<<<dim3(FO), T, 0, stream>>>(h1, FO, NTOT, nullptr, (float)NTOT, mC, vC);
        bn_apply<<<g1d(NTOT * FO), T, 0, stream>>>(h1, f2n, FO, NTOT, mC, vC, g2[l], be2[l], 1);

        // ---- edge weights + degrees, then clipped aggregation ----
        edge_w_deg<<<dim3(NTOT / 8), 256, 0, stream>>>(f1n, f2n, idx, wbuf, deg, FO);
        float* outbuf = (l == 2) ? hout : f1n;   // f1n free after edge_w_deg
        aggregate<<<dim3(NTOT / 8), 256, 0, stream>>>(xp, wbuf, deg, idx, outbuf, FO);

        // ---- dense concat + PReLU + BN (layers 0,1) ----
        if (l < 2) {
            const int Cn = F + FO;
            concat_prelu<<<g1d(NTOT * Cn), T, 0, stream>>>(hcur, F, outbuf, FO, act_a[l], hnext);
            colstats<<<dim3(Cn), T, 0, stream>>>(hnext, Cn, NTOT, nullptr, (float)NTOT, mC, vC);
            bn_apply<<<g1d(NTOT * Cn), T, 0, stream>>>(hnext, hnext, Cn, NTOT, mC, vC,
                                                       act_g[l], act_be[l], 0);
            hcur = hnext;
            hnext = (hnext == hA) ? hB : hA;
        }
    }

    // ---- loss: mse = mean((h - y)^2); total = mse ----
    zero_f32<<<dim3(1), 32, 0, stream>>>(acc, 1);
    mse_partial<<<dim3(64), 256, 0, stream>>>(hout, y, NTOT * 6, acc);
    mse_final<<<dim3(1), 32, 0, stream>>>(acc, hout + (size_t)NTOT * 6, 1.0f / (NTOT * 6));
}
